// AdaptiveSpectralDNA_82849919139907
// MI455X (gfx1250) — matrix operand order repaired and checked by reading, not compiled, not run on hardware
//
#include <hip/hip_runtime.h>

typedef __attribute__((ext_vector_type(16))) _Float16 v16h;
typedef __attribute__((ext_vector_type(8)))  _Float16 v8h;
typedef __attribute__((ext_vector_type(8)))  float    v8f;
typedef __attribute__((ext_vector_type(4)))  unsigned v4u;
typedef __attribute__((ext_vector_type(8)))  int      v8i;
typedef __attribute__((ext_vector_type(4)))  int      v4i;

#define HIDDEN        256
#define NLAYERS       4
#define ROWS_PER_WAVE 32                         // two 16-row WMMA M-tiles per wave
#define NWAVES        8
#define ROWS_PER_BLK  (ROWS_PER_WAVE * NWAVES)   // 256
#define LAYER_HALF    (HIDDEN * HIDDEN)          // 65536 f16 per layer
// LDS layout (dynamic shared)
#define WBUF_OFF 0
#define ACT_OFF  131072                                            // 128 KB weights
#define OM_OFF   (ACT_OFF + NWAVES * ROWS_PER_WAVE * HIDDEN * 2)   // +128 KB act
#define CB_OFF   (OM_OFF + NWAVES * ROWS_PER_WAVE * 4)             // +1 KB omega
#define SMEM_BYTES (CB_OFF + NWAVES * ROWS_PER_WAVE * 16)          // +4 KB coords

// ---------------------------------------------------------------------------
// Prep: fp32 wh[4][256][256] -> f16, pre-swizzled into WMMA B-fragment order:
// wt[l][kc][nt][lane][e], where K = kc*32 + e + (lane>=16 ? 16 : 0),
//                         N = nt*16 + (lane & 15).
// Each lane's 16 halves (32 B) are then contiguous in LDS/global.
// ---------------------------------------------------------------------------
__global__ void siren_prep(const float* __restrict__ wh, _Float16* __restrict__ wt) {
  int idx  = blockIdx.x * blockDim.x + threadIdx.x;   // 262144 total
  int e    = idx & 15;
  int lane = (idx >> 4) & 31;
  int nt   = (idx >> 9) & 15;
  int kc   = (idx >> 13) & 7;
  int l    = (idx >> 16) & 3;
  int K = kc * 32 + e + ((lane >= 16) ? 16 : 0);
  int Nn = nt * 16 + (lane & 15);
  wt[idx] = (_Float16)wh[(l * 256 + K) * 256 + Nn];
}

// ---------------------------------------------------------------------------
// TDM: stage one layer's 128 KB tiled weight block into LDS (1-D descriptor,
// data_size = 8 B, 16384 elements). Issued by wave 0 only; 6-arg builtin.
// ---------------------------------------------------------------------------
#if defined(__gfx1250__) && __has_builtin(__builtin_amdgcn_tensor_load_to_lds)
#define USE_TDM 1
__device__ __forceinline__ void tdm_load_layer(const _Float16* gsrc, unsigned lds_off) {
  unsigned long long ga = (unsigned long long)(size_t)gsrc;
  v4u g0;
  g0[0] = 1u;                                           // count = 1 valid descriptor
  g0[1] = lds_off;                                      // lds_addr
  g0[2] = (unsigned)ga;                                 // global_addr[31:0]
  g0[3] = ((unsigned)(ga >> 32) & 0x01FFFFFFu)          // global_addr[56:32]
          | (2u << 30);                                 // type = 2 ("image")
  v8i g1;
  g1[0] = 3 << 16;                                      // data_size = 3 (8 bytes)
  g1[1] = (int)(16384u << 16);                          // tensor_dim0 = 16384 (lo16)
  g1[2] = (int)(1u << 16);                              // tensor_dim1 = 1
  g1[3] = (int)(16384u << 16);                          // tile_dim0 = 16384
  g1[4] = 0;                                            // tile_dim1/2 unused
  g1[5] = 16384;                                        // tensor_dim0_stride
  g1[6] = 0;
  g1[7] = 0;
  v4i z4 = {0, 0, 0, 0};
  v8i z8 = {0, 0, 0, 0, 0, 0, 0, 0};
  __builtin_amdgcn_tensor_load_to_lds(g0, g1, z4, z4, z8, 0);
}
#else
#define USE_TDM 0
#endif

// ---------------------------------------------------------------------------
// Fused SIREN forward. 256 threads = 8 waves; each wave owns 32 rows
// (two 16-row M-tiles), so every B fragment read from LDS feeds two WMMAs.
// ---------------------------------------------------------------------------
__global__ __launch_bounds__(256)
void siren_fused(const float* __restrict__ coords,
                 const float* __restrict__ ow1, const float* __restrict__ ob1,
                 const float* __restrict__ ow2, const float* __restrict__ ob2,
                 const float* __restrict__ w0,  const float* __restrict__ b0,
                 const float* __restrict__ bh,  const float* __restrict__ wf,
                 const float* __restrict__ bf,
                 const _Float16* __restrict__ wt,
                 float* __restrict__ out)
{
  extern __shared__ char smem[];
  _Float16* wbuf = (_Float16*)(smem + WBUF_OFF);   // [8][16][32][16] f16 tiles
  _Float16* act  = (_Float16*)(smem + ACT_OFF);    // [wave][32][256] f16
  float*    om   = (float*)(smem + OM_OFF);        // [wave][32]
  float*    cb   = (float*)(smem + CB_OFF);        // [wave][32][4]

  const int tid  = threadIdx.x;
  const int wave = tid >> 5;
  const int lane = tid & 31;
  const int mrow = lane & 15;
  const int koff = (lane >> 4) << 3;               // 0 or 8 (lane-half select)
  const int row0 = blockIdx.x * ROWS_PER_BLK + wave * ROWS_PER_WAVE;

  _Float16* actw = act + wave * (ROWS_PER_WAVE * HIDDEN);
  float*    omw  = om + wave * ROWS_PER_WAVE;
  float*    cbw  = cb + wave * (ROWS_PER_WAVE * 4);

#if USE_TDM
  if (wave == 0) tdm_load_layer(wt, (unsigned)(size_t)wbuf);   // prefetch layer 0
#endif

  // ---- omega predictor (4 -> 64 relu -> 1 sigmoid); one row per lane ----
  {
    int g = row0 + lane;
    float c0 = coords[g * 4 + 0], c1 = coords[g * 4 + 1];
    float c2 = coords[g * 4 + 2], c3 = coords[g * 4 + 3];
    float s = ob2[0];
    #pragma unroll 8
    for (int j = 0; j < 64; ++j) {
      float h = fmaf(c0, ow1[j],
                fmaf(c1, ow1[64 + j],
                fmaf(c2, ow1[128 + j],
                fmaf(c3, ow1[192 + j], ob1[j]))));
      h = fmaxf(h, 0.f);
      s = fmaf(h, ow2[j], s);
    }
    float omega = 10.f + 90.f / (1.f + __expf(-s));
    omw[lane] = omega;
    cbw[lane * 4 + 0] = c0; cbw[lane * 4 + 1] = c1;
    cbw[lane * 4 + 2] = c2; cbw[lane * 4 + 3] = c3;
  }

  // ---- layer 0: x = sin(omega * (coords @ w0 + b0)), K=4 -> plain VALU ----
  #pragma unroll 4
  for (int t = 0; t < 256; ++t) {
    int idx = t * 32 + lane;                 // 32*256 outputs per wave
    int r = idx >> 8, c = idx & 255;
    float v = fmaf(cbw[r * 4 + 0], w0[c],
              fmaf(cbw[r * 4 + 1], w0[256 + c],
              fmaf(cbw[r * 4 + 2], w0[512 + c],
              fmaf(cbw[r * 4 + 3], w0[768 + c], b0[c]))));
    actw[r * 256 + c] = (_Float16)__sinf(omw[r] * v);
  }

  // omega values for this lane's D-fragment rows of both M-tiles
  float om0[8], om1[8];
  #pragma unroll
  for (int v = 0; v < 8; ++v) {
    om0[v] = omw[v + koff];
    om1[v] = omw[16 + v + koff];
  }

  // ---- hidden layers: x = sin(omega * (x @ Wl + bl)) via WMMA ----
  for (int l = 0; l < NLAYERS; ++l) {
#if USE_TDM
    if (wave == 0) __builtin_amdgcn_s_wait_tensorcnt(0);
#else
    for (int i = tid * 8; i < LAYER_HALF; i += 256 * 8)
      *(float4*)(wbuf + i) = *(const float4*)(wt + l * LAYER_HALF + i);
#endif
    __syncthreads();                         // weights staged; act writes visible

    // A fragments for both 16x256 input tiles (register resident: 128 VGPRs)
    v16h a0[8], a1[8];
    #pragma unroll
    for (int kc = 0; kc < 8; ++kc) {
      const _Float16* p0 = actw + mrow * 256 + kc * 32 + 2 * koff;
      const _Float16* p1 = p0 + 16 * 256;
      union { v16h v; v8h h[2]; } u0, u1;
      u0.h[0] = *(const v8h*)(p0);           // K = base + 2*koff + 0..7
      u0.h[1] = *(const v8h*)(p0 + 8);       // K = base + 2*koff + 8..15
      u1.h[0] = *(const v8h*)(p1);
      u1.h[1] = *(const v8h*)(p1 + 8);
      a0[kc] = u0.v;
      a1[kc] = u1.v;
    }

    for (int nt = 0; nt < 16; ++nt) {
      v8f acc0 = {}, acc1 = {};
      #pragma unroll
      for (int kc = 0; kc < 8; ++kc) {
        // one B fragment feeds two independent WMMA chains
        const v16h b = *(const v16h*)(wbuf + (((kc * 16 + nt) * 32) + lane) * 16);
        acc0 = __builtin_amdgcn_wmma_f32_16x16x32_f16(
                   false, a0[kc], false, b, (short)0, acc0, false, false);
        acc1 = __builtin_amdgcn_wmma_f32_16x16x32_f16(
                   false, a1[kc], false, b, (short)0, acc1, false, false);
      }
      float bn = bh[l * 256 + nt * 16 + mrow];   // bias per output column
      #pragma unroll
      for (int v = 0; v < 8; ++v) {
        float y0 = __sinf(om0[v] * (acc0[v] + bn));
        float y1 = __sinf(om1[v] * (acc1[v] + bn));
        actw[(v + koff) * 256 + nt * 16 + mrow]        = (_Float16)y0;
        actw[(16 + v + koff) * 256 + nt * 16 + mrow]   = (_Float16)y1;
      }
    }
    __syncthreads();                         // everyone done reading wbuf
#if USE_TDM
    if (wave == 0 && l + 1 < NLAYERS)
      tdm_load_layer(wt + (l + 1) * LAYER_HALF, (unsigned)(size_t)wbuf);
#endif
  }

  // ---- final layer: out = x @ wf + bf; one row per lane, rotated chunks ----
  {
    float accf = 0.f;
    const _Float16* p = actw + lane * 256;
    #pragma unroll 4
    for (int c = 0; c < 32; ++c) {
      int cc = (lane + c) & 31;              // rotate start: avoid 32-way LDS conflict
      v8h h = *(const v8h*)(p + cc * 8);
      #pragma unroll
      for (int j = 0; j < 8; ++j)
        accf = fmaf((float)h[j], wf[cc * 8 + j], accf);
    }
    out[row0 + lane] = accf + bf[0];
  }
}

// ---------------------------------------------------------------------------
extern "C" void kernel_launch(void* const* d_in, const int* in_sizes, int n_in,
                              void* d_out, int out_size, void* d_ws, size_t ws_size,
                              hipStream_t stream) {
  (void)n_in; (void)ws_size; (void)out_size;
  const float* coords = (const float*)d_in[0];
  const float* ow1    = (const float*)d_in[1];
  const float* ob1    = (const float*)d_in[2];
  const float* ow2    = (const float*)d_in[3];
  const float* ob2    = (const float*)d_in[4];
  const float* w0     = (const float*)d_in[5];
  const float* b0     = (const float*)d_in[6];
  const float* wh     = (const float*)d_in[7];
  const float* bh     = (const float*)d_in[8];
  const float* wf     = (const float*)d_in[9];
  const float* bf     = (const float*)d_in[10];

  _Float16* wt = (_Float16*)d_ws;            // 4 * 128 KB pre-swizzled f16 weights

  const int nrows = in_sizes[0] / 4;         // 524288
  siren_prep<<<(NLAYERS * LAYER_HALF) / 256, 256, 0, stream>>>(wh, wt);
  siren_fused<<<nrows / ROWS_PER_BLK, 256, SMEM_BYTES, stream>>>(
      coords, ow1, ob1, ow2, ob2, w0, b0, bh, wf, bf, wt, (float*)d_out);
}
